// LSTMPredictor_81372450390636
// MI455X (gfx1250) — compile-verified, matrix operand order checked by reading, not run on hardware
//
#include <hip/hip_runtime.h>

#define B_   1024
#define T_   512
#define D_   8
#define H_   128
#define G_   512          // 4*H gates
#define NT_  32           // G/16 gate column tiles
#define KF_  4            // H/32 K-fragments for f16 wmma
#define BT_  16           // batch rows per workgroup (one M tile)
#define TPB_ 256          // 8 waves (wave32)

typedef __attribute__((ext_vector_type(16))) _Float16 v16h;
typedef __attribute__((ext_vector_type(8)))  float    v8f;
typedef __attribute__((ext_vector_type(4)))  float    v4f;

union Frag16 { v16h v; v4f f4[2]; };

__device__ __forceinline__ float sigf(float x) {
  return __builtin_amdgcn_rcpf(1.0f + __expf(-x));
}
__device__ __forceinline__ float tanh_fast(float x) {
  // tanh(x) = 1 - 2/(exp(2x)+1)
  return 1.0f - 2.0f * __builtin_amdgcn_rcpf(1.0f + __expf(2.0f * x));
}

// Pack a (G x Ksrc) f32 row-major weight matrix into f16 WMMA B-operand
// fragments in LDS, zero-padding K up to Kpad (multiple of 32).
// Per 16x16x32 block (nt,kf): 32 lanes x 16 halfs (32B per lane, contiguous).
// lane l, slot s -> B[k][n] with n = nt*16 + (l&15),
//                  k = kf*32 + (s>>3)*16 + (l>>4)*8 + (s&7)
__device__ __forceinline__ void pack_b(const float* __restrict__ W,
                                       _Float16* __restrict__ pk,
                                       int Kpad, int Ksrc, int tid) {
  const int KF  = Kpad >> 5;
  const int tot = NT_ * KF * 32;
  for (int idx = tid; idx < tot; idx += TPB_) {
    const int l   = idx & 31;
    const int blk = idx >> 5;
    const int nt  = blk / KF;
    const int kf  = blk % KF;
    const int n   = nt * 16 + (l & 15);
    const int lh  = (l >> 4) * 8;
#pragma unroll
    for (int s = 0; s < 16; ++s) {
      const int k = kf * 32 + ((s >> 3) << 4) + lh + (s & 7);
      pk[idx * 16 + s] = (k < Ksrc) ? (_Float16)W[n * Ksrc + k] : (_Float16)0.0f;
    }
  }
}

// ---------------- Kernel 1: LSTM layer 0, writes h0[B,T,H] as f16 to ws ----
__global__ void __launch_bounds__(TPB_)
lstm_layer0_kernel(const float* __restrict__ x,
                   const float* __restrict__ Wih, const float* __restrict__ Whh,
                   const float* __restrict__ bih, const float* __restrict__ bhh,
                   _Float16* __restrict__ h0out)
{
  extern __shared__ __align__(32) char smem[];
  _Float16* whh_pk = (_Float16*)(smem);                 // 131072 B
  _Float16* wih_pk = (_Float16*)(smem + 131072);        //  32768 B (Kpad=32)
  float*    biasS  = (float*)(smem + 163840);           //   2048 B
  _Float16* xbuf   = (_Float16*)(smem + 165888);        //   1024 B (16x32 f16)
  _Float16* hbuf   = (_Float16*)(smem + 166912);        //   4096 B (16x128 f16)
  // total 171008 B

  const int tid  = threadIdx.x;
  const int wv   = tid >> 5;
  const int lane = tid & 31;
  const int hlf  = lane >> 4;
  const int ln16 = lane & 15;
  const int b0   = blockIdx.x * BT_;

  pack_b(Whh, whh_pk, 128, 128, tid);
  pack_b(Wih, wih_pk, 32, 8, tid);
  for (int i = tid; i < G_; i += TPB_)       biasS[i] = bih[i] + bhh[i];
  for (int i = tid; i < BT_ * 32; i += TPB_) xbuf[i]  = (_Float16)0.0f;
  for (int i = tid; i < BT_ * H_; i += TPB_) hbuf[i]  = (_Float16)0.0f;
  __syncthreads();

  const int ntg[4] = { wv, 8 + wv, 16 + wv, 24 + wv };
  const int hcol = wv * 16 + ln16;
  const Frag16* whhF = (const Frag16*)whh_pk;
  const Frag16* wihF = (const Frag16*)wih_pk;
  v8f c = {0.f, 0.f, 0.f, 0.f, 0.f, 0.f, 0.f, 0.f};

  for (int t = 0; t < T_; ++t) {
    if (tid < BT_ * D_) {
      const int r = tid >> 3, d = tid & 7;
      xbuf[r * 32 + d] = (_Float16)x[((size_t)(b0 + r) * T_ + t) * D_ + d];
    }
    __syncthreads();

    v8f acc[4];
#pragma unroll
    for (int g = 0; g < 4; ++g) {
      const float bv = biasS[g * H_ + hcol];
      v8f bvv = {bv, bv, bv, bv, bv, bv, bv, bv};
      acc[g] = bvv;
    }
    // input projection: x_t (16x8 zero-padded to K=32) @ Wih^T
    {
      Frag16 a;
      const _Float16* p = xbuf + ln16 * 32 + hlf * 8;
      a.f4[0] = *(const v4f*)(p);
      a.f4[1] = *(const v4f*)(p + 16);
#pragma unroll
      for (int g = 0; g < 4; ++g) {
        const Frag16 b = wihF[ntg[g] * 32 + lane];
        acc[g] = __builtin_amdgcn_wmma_f32_16x16x32_f16(
            false, a.v, false, b.v, (short)0, acc[g], false, false);
      }
    }
    // recurrent: h_{t-1} (16x128) @ Whh^T, K=128 in 4 fragments
#pragma unroll
    for (int kf = 0; kf < KF_; ++kf) {
      Frag16 a;
      const _Float16* p = hbuf + ln16 * H_ + kf * 32 + hlf * 8;
      a.f4[0] = *(const v4f*)(p);
      a.f4[1] = *(const v4f*)(p + 16);
#pragma unroll
      for (int g = 0; g < 4; ++g) {
        const Frag16 b = whhF[(ntg[g] * KF_ + kf) * 32 + lane];
        acc[g] = __builtin_amdgcn_wmma_f32_16x16x32_f16(
            false, a.v, false, b.v, (short)0, acc[g], false, false);
      }
    }
    __syncthreads();
    // cell update; wave wv owns h columns [wv*16, wv*16+16)
#pragma unroll
    for (int r = 0; r < 8; ++r) {
      const float ig = sigf(acc[0][r]);
      const float fg = sigf(acc[1][r]);
      const float gg = tanh_fast(acc[2][r]);
      const float og = sigf(acc[3][r]);
      const float cv = fg * c[r] + ig * gg;
      c[r] = cv;
      const float hv = og * tanh_fast(cv);
      const int row = r + hlf * 8;
      const _Float16 hh = (_Float16)hv;
      hbuf[row * H_ + hcol] = hh;
      h0out[((size_t)(b0 + row) * T_ + t) * H_ + hcol] = hh;
    }
  }
}

// ---------------- Kernel 2: LSTM layer 1 + FC head + softmax ---------------
__global__ void __launch_bounds__(TPB_)
lstm_layer1_head_kernel(const _Float16* __restrict__ h0,
                        const float* __restrict__ Wih, const float* __restrict__ Whh,
                        const float* __restrict__ bih, const float* __restrict__ bhh,
                        const float* __restrict__ fc1w, const float* __restrict__ fc1b,
                        const float* __restrict__ fc2w, const float* __restrict__ fc2b,
                        float* __restrict__ out)
{
  extern __shared__ __align__(32) char smem[];
  _Float16* wih_pk = (_Float16*)(smem);            // 131072 B
  _Float16* whh_pk = (_Float16*)(smem + 131072);   // 131072 B
  float*    biasS  = (float*)(smem + 262144);      //   2048 B
  _Float16* h0buf  = (_Float16*)(smem + 264192);   //   4096 B
  _Float16* h1buf  = (_Float16*)(smem + 268288);   //   4096 B
  float*    hf     = (float*)(smem + 272384);      //   8192 B (final h, f32)
  float*    zbuf   = (float*)(smem + 280576);      //   2048 B
  // total 282624 B  (< 320KB WGP LDS)

  const int tid  = threadIdx.x;
  const int wv   = tid >> 5;
  const int lane = tid & 31;
  const int hlf  = lane >> 4;
  const int ln16 = lane & 15;
  const int b0   = blockIdx.x * BT_;

  pack_b(Wih, wih_pk, 128, 128, tid);
  pack_b(Whh, whh_pk, 128, 128, tid);
  for (int i = tid; i < G_; i += TPB_)       biasS[i] = bih[i] + bhh[i];
  for (int i = tid; i < BT_ * H_; i += TPB_) h1buf[i] = (_Float16)0.0f;
  __syncthreads();

  const int ntg[4] = { wv, 8 + wv, 16 + wv, 24 + wv };
  const int hcol = wv * 16 + ln16;
  const Frag16* wihF = (const Frag16*)wih_pk;
  const Frag16* whhF = (const Frag16*)whh_pk;
  v8f c = {0.f, 0.f, 0.f, 0.f, 0.f, 0.f, 0.f, 0.f};

  for (int t = 0; t < T_; ++t) {
    // coalesced stage of h0_t tile (16 x 128 f16), one b128 per thread
    {
      const int row = tid >> 4, ch = tid & 15;
      const size_t src = ((size_t)(b0 + row) * T_ + t) * H_ + ch * 8;
      *(v4f*)&h0buf[row * H_ + ch * 8] = *(const v4f*)(h0 + src);
      if (t + 1 < T_) __builtin_prefetch(h0 + src + H_, 0, 1);
    }
    __syncthreads();

    v8f acc[4];
#pragma unroll
    for (int g = 0; g < 4; ++g) {
      const float bv = biasS[g * H_ + hcol];
      v8f bvv = {bv, bv, bv, bv, bv, bv, bv, bv};
      acc[g] = bvv;
    }
#pragma unroll
    for (int kf = 0; kf < KF_; ++kf) {
      Frag16 a0, a1;
      const _Float16* p0 = h0buf + ln16 * H_ + kf * 32 + hlf * 8;
      a0.f4[0] = *(const v4f*)(p0);
      a0.f4[1] = *(const v4f*)(p0 + 16);
      const _Float16* p1 = h1buf + ln16 * H_ + kf * 32 + hlf * 8;
      a1.f4[0] = *(const v4f*)(p1);
      a1.f4[1] = *(const v4f*)(p1 + 16);
#pragma unroll
      for (int g = 0; g < 4; ++g) {
        const Frag16 bi = wihF[(ntg[g] * KF_ + kf) * 32 + lane];
        acc[g] = __builtin_amdgcn_wmma_f32_16x16x32_f16(
            false, a0.v, false, bi.v, (short)0, acc[g], false, false);
        const Frag16 bh = whhF[(ntg[g] * KF_ + kf) * 32 + lane];
        acc[g] = __builtin_amdgcn_wmma_f32_16x16x32_f16(
            false, a1.v, false, bh.v, (short)0, acc[g], false, false);
      }
    }
    __syncthreads();
#pragma unroll
    for (int r = 0; r < 8; ++r) {
      const float ig = sigf(acc[0][r]);
      const float fg = sigf(acc[1][r]);
      const float gg = tanh_fast(acc[2][r]);
      const float og = sigf(acc[3][r]);
      const float cv = fg * c[r] + ig * gg;
      c[r] = cv;
      const float hv = og * tanh_fast(cv);
      const int row = r + hlf * 8;
      h1buf[row * H_ + hcol] = (_Float16)hv;
      if (t == T_ - 1) hf[row * H_ + hcol] = hv;
    }
  }
  __syncthreads();

  // FC1 (128->32) + ReLU : 16 rows x 32 cols, 2 outputs per thread
  {
    const int m  = tid >> 4;
    const int j0 = tid & 15;
#pragma unroll
    for (int jj = 0; jj < 2; ++jj) {
      const int j = j0 + jj * 16;
      float s = fc1b[j];
      for (int k = 0; k < H_; ++k) s = fmaf(hf[m * H_ + k], fc1w[j * H_ + k], s);
      zbuf[m * 32 + j] = fmaxf(s, 0.0f);
    }
  }
  __syncthreads();

  // FC2 (32->3) + softmax, one thread per batch row
  if (tid < BT_) {
    const int m = tid;
    float lg[3];
#pragma unroll
    for (int o = 0; o < 3; ++o) {
      float s = fc2b[o];
      for (int j = 0; j < 32; ++j) s = fmaf(zbuf[m * 32 + j], fc2w[o * 32 + j], s);
      lg[o] = s;
    }
    const float mx = fmaxf(lg[0], fmaxf(lg[1], lg[2]));
    const float e0 = __expf(lg[0] - mx);
    const float e1 = __expf(lg[1] - mx);
    const float e2 = __expf(lg[2] - mx);
    const float inv = __builtin_amdgcn_rcpf(e0 + e1 + e2);
    out[(size_t)(b0 + m) * 3 + 0] = e0 * inv;
    out[(size_t)(b0 + m) * 3 + 1] = e1 * inv;
    out[(size_t)(b0 + m) * 3 + 2] = e2 * inv;
  }
}

extern "C" void kernel_launch(void* const* d_in, const int* in_sizes, int n_in,
                              void* d_out, int out_size, void* d_ws, size_t ws_size,
                              hipStream_t stream) {
  (void)in_sizes; (void)n_in; (void)out_size; (void)ws_size;
  const float* x    = (const float*)d_in[0];
  const float* Wih0 = (const float*)d_in[1];
  const float* Whh0 = (const float*)d_in[2];
  const float* bih0 = (const float*)d_in[3];
  const float* bhh0 = (const float*)d_in[4];
  const float* Wih1 = (const float*)d_in[5];
  const float* Whh1 = (const float*)d_in[6];
  const float* bih1 = (const float*)d_in[7];
  const float* bhh1 = (const float*)d_in[8];
  const float* fc1w = (const float*)d_in[9];
  const float* fc1b = (const float*)d_in[10];
  const float* fc2w = (const float*)d_in[11];
  const float* fc2b = (const float*)d_in[12];

  // workspace: h0 activations as f16, B*T*H*2 = 128 MB
  _Float16* h0ws = (_Float16*)d_ws;

  const int lds1 = 171008;
  const int lds2 = 282624;
  hipFuncSetAttribute(reinterpret_cast<const void*>(lstm_layer0_kernel),
                      hipFuncAttributeMaxDynamicSharedMemorySize, lds1);
  hipFuncSetAttribute(reinterpret_cast<const void*>(lstm_layer1_head_kernel),
                      hipFuncAttributeMaxDynamicSharedMemorySize, lds2);

  lstm_layer0_kernel<<<dim3(B_ / BT_), dim3(TPB_), lds1, stream>>>(
      x, Wih0, Whh0, bih0, bhh0, h0ws);
  lstm_layer1_head_kernel<<<dim3(B_ / BT_), dim3(TPB_), lds2, stream>>>(
      h0ws, Wih1, Whh1, bih1, bhh1, fc1w, fc1b, fc2w, fc2b, (float*)d_out);
}
